// nystromAttention_52029233823830
// MI455X (gfx1250) — matrix operand order repaired and checked by reading, not compiled, not run on hardware
//
#include <hip/hip_runtime.h>
#include <hip/hip_bf16.h>
#include <stdint.h>

// ---------------- problem constants ----------------
#define B_    4
#define N_    4096
#define DIM_  1024
#define H_    16
#define HD_   64
#define BN_   (B_ * N_)     // 16384
#define K3_   (3 * DIM_)    // 3072
#define BH_   (B_ * H_)     // 64

// ---------------- WMMA helpers (gfx1250, wave32) ----------------
typedef __attribute__((ext_vector_type(16))) __bf16 v16bf;
typedef __attribute__((ext_vector_type(8)))  float  v8f;

union BF16x16 { v16bf v; uint4 q[2]; };

__device__ __forceinline__ v8f v8f_zero() {
    v8f z;
#pragma unroll
    for (int e = 0; e < 8; ++e) z[e] = 0.f;
    return z;
}

// A fragment: 16x32 bf16, row-major in LDS with `stride` bf16 per row.
// Lanes 0-15: row=lane, elems 0..7 -> K 0..7, elems 8..15 -> K 16..23
// Lanes 16-31: row=lane-16, elems 0..7 -> K 8..15, elems 8..15 -> K 24..31
__device__ __forceinline__ v16bf ldsA16x32(const __bf16* base, int stride) {
    const int lane = threadIdx.x & 31;
    const int row  = lane & 15;
    const int kb   = (lane >> 4) << 3;   // 0 or 8
    BF16x16 u;
    const __bf16* r = base + row * stride;
    u.q[0] = *(const uint4*)(r + kb);
    u.q[1] = *(const uint4*)(r + 16 + kb);
    return u.v;
}

// B fragment: 32x16 bf16 where LDS holds B transposed [N][K] (stride bf16/row).
// Lanes 0-15: col n=lane, elems 0..7 -> K 0..7, elems 8..15 -> K 16..23
// Lanes 16-31: col n=lane-16, elems 0..7 -> K 8..15, elems 8..15 -> K 24..31
__device__ __forceinline__ v16bf ldsB32x16(const __bf16* base, int stride) {
    const int lane = threadIdx.x & 31;
    const int n    = lane & 15;
    const int kb   = (lane >> 4) << 3;   // 0 or 8
    BF16x16 u;
    const __bf16* r = base + n * stride + kb;
    u.q[0] = *(const uint4*)(r);
    u.q[1] = *(const uint4*)(r + 16);
    return u.v;
}

__device__ __forceinline__ v8f wmma_bf16(v16bf a, v16bf b, v8f c) {
    return __builtin_amdgcn_wmma_f32_16x16x32_bf16(
        false, a, false, b, (short)0, c, false, false);
}

// gfx1250 async DMA: global -> LDS, 16 bytes per lane, tracked by ASYNCcnt.
// ldsOff = low 32 bits of the generic shared-memory address (flat->LDS mapping
// truncates to addr[31:0], ISA 10.2), gp = per-lane global address.
__device__ __forceinline__ void async_ld_b128(uint32_t ldsOff, const __bf16* gp) {
    asm volatile("global_load_async_to_lds_b128 %0, %1, off"
                 : : "v"(ldsOff), "v"(gp) : "memory");
}

// ---------------- conversion kernels ----------------
__global__ void k_cvt(const float* __restrict__ src, __bf16* __restrict__ dst, int n) {
    int i = blockIdx.x * blockDim.x + threadIdx.x;
    if (i < n) dst[i] = (__bf16)src[i];
}

// src [rows][cols] fp32 -> dst [cols][rows] bf16
__global__ void k_cvt_T(const float* __restrict__ src, __bf16* __restrict__ dst,
                        int rows, int cols) {
    int o = blockIdx.x * blockDim.x + threadIdx.x;
    if (o >= rows * cols) return;
    int oc = o / rows;      // 0..cols-1
    int ok = o - oc * rows; // 0..rows-1
    dst[o] = (__bf16)src[(size_t)ok * cols + oc];
}

// ---------------- big GEMM: C[16384 x Ntot] = A[16384 x K] * B[K x Ntot] ----
// Bt is pre-transposed [Ntot][K] bf16.  mode 0: QKV epilogue (split heads,
// scale Q by 1/sqrt(HD)=0.125).  mode 1: write fp32 to outF [BN_ x DIM_].
// Double-buffered LDS fed by GLOBAL_LOAD_ASYNC_TO_LDS_B128 (ASYNCcnt).
__global__ __launch_bounds__(256)
void k_gemm128(const __bf16* __restrict__ A, const __bf16* __restrict__ Bt,
               int Kdim, int mode,
               __bf16* __restrict__ Qb, __bf16* __restrict__ Kb,
               __bf16* __restrict__ Vb, float* __restrict__ outF) {
    __shared__ alignas(16) __bf16 As[2][128 * 32];
    __shared__ alignas(16) __bf16 Bs[2][128 * 32];
    const int tid  = threadIdx.x;
    const int lane = tid & 31, wave = tid >> 5;
    const int m0 = (wave >> 2) * 64;     // wave tile 64x32 inside 128x128 block
    const int n0 = (wave & 3) * 32;
    const int rowBase = blockIdx.y * 128;
    const int colBase = blockIdx.x * 128;

    // Each thread owns 4 fixed 16B copy slots per tile (512 chunks / 256 thr
    // for A, same for B -> 2 A-slots + 2 B-slots = 4 async ops per issue).
    const int c0 = tid, c1 = tid + 256;
    const int r0 = c0 >> 2, o0 = (c0 & 3) << 3;
    const int r1 = c1 >> 2, o1 = (c1 & 3) << 3;
    const __bf16* gA0 = A  + (size_t)(rowBase + r0) * Kdim + o0;
    const __bf16* gA1 = A  + (size_t)(rowBase + r1) * Kdim + o1;
    const __bf16* gB0 = Bt + (size_t)(colBase + r0) * Kdim + o0;
    const __bf16* gB1 = Bt + (size_t)(colBase + r1) * Kdim + o1;
    uint32_t la0[2], la1[2], lb0[2], lb1[2];
#pragma unroll
    for (int b = 0; b < 2; ++b) {
        la0[b] = (uint32_t)(uintptr_t)&As[b][r0 * 32 + o0];
        la1[b] = (uint32_t)(uintptr_t)&As[b][r1 * 32 + o1];
        lb0[b] = (uint32_t)(uintptr_t)&Bs[b][r0 * 32 + o0];
        lb1[b] = (uint32_t)(uintptr_t)&Bs[b][r1 * 32 + o1];
    }

    v8f acc[4][2];
#pragma unroll
    for (int i = 0; i < 4; ++i)
#pragma unroll
        for (int j = 0; j < 2; ++j) acc[i][j] = v8f_zero();

    // prologue: fill buffer 0
    async_ld_b128(la0[0], gA0);
    async_ld_b128(la1[0], gA1);
    async_ld_b128(lb0[0], gB0);
    async_ld_b128(lb1[0], gB1);

    const int nk = Kdim >> 5;
    for (int kt = 0; kt < nk; ++kt) {
        const int buf = kt & 1;
        if (kt + 1 < nk) {
            const int k0 = (kt + 1) << 5;
            async_ld_b128(la0[buf ^ 1], gA0 + k0);
            async_ld_b128(la1[buf ^ 1], gA1 + k0);
            async_ld_b128(lb0[buf ^ 1], gB0 + k0);
            async_ld_b128(lb1[buf ^ 1], gB1 + k0);
            // 8 async ops in flight; first 4 (current buf) complete in order.
            asm volatile("s_wait_asynccnt 0x4" ::: "memory");
        } else {
            asm volatile("s_wait_asynccnt 0x0" ::: "memory");
        }
        __syncthreads();
        v16bf af[4], bfv[2];
#pragma unroll
        for (int i = 0; i < 4; ++i) af[i]  = ldsA16x32(&As[buf][(m0 + i * 16) * 32], 32);
#pragma unroll
        for (int j = 0; j < 2; ++j) bfv[j] = ldsB32x16(&Bs[buf][(n0 + j * 16) * 32], 32);
#pragma unroll
        for (int i = 0; i < 4; ++i)
#pragma unroll
            for (int j = 0; j < 2; ++j) acc[i][j] = wmma_bf16(af[i], bfv[j], acc[i][j]);
        __syncthreads();   // protects buf^1 (written by next iteration's issue)
    }

    const int er = (lane >> 4) << 3;   // acc row offset within 16x16 tile
    const int ec = lane & 15;          // acc col
#pragma unroll
    for (int i = 0; i < 4; ++i)
#pragma unroll
        for (int j = 0; j < 2; ++j)
#pragma unroll
            for (int e = 0; e < 8; ++e) {
                int gm = rowBase + m0 + i * 16 + er + e;
                int gn = colBase + n0 + j * 16 + ec;
                float v = acc[i][j][e];
                if (mode == 0) {
                    int which = gn >> 10;             // 0=Q 1=K 2=V
                    int h = (gn & 1023) >> 6;
                    int d = gn & 63;
                    int b = gm >> 12, n = gm & 4095;
                    size_t off = ((size_t)(b * H_ + h) * N_ + n) * HD_ + d;
                    if (which == 0)      Qb[off] = (__bf16)(v * 0.125f);
                    else if (which == 1) Kb[off] = (__bf16)v;
                    else                 Vb[off] = (__bf16)v;
                } else {
                    outF[(size_t)gm * DIM_ + gn] = v;
                }
            }
}

// ---------------- landmark means: Q_l, K_l [BH][64][64] ----------------
__global__ void k_landmark(const __bf16* __restrict__ Qb, const __bf16* __restrict__ Kb,
                           __bf16* __restrict__ Qlb, __bf16* __restrict__ Klb) {
    int idx = blockIdx.x * blockDim.x + threadIdx.x;  // 2*64*64*64 = 524288
    int sel = idx >> 18;
    int bh  = (idx >> 12) & 63, m = (idx >> 6) & 63, d = idx & 63;
    const __bf16* src = (sel ? Kb : Qb) + ((size_t)bh * N_ + m * 64) * HD_ + d;
    float s = 0.f;
#pragma unroll 8
    for (int i = 0; i < 64; ++i) s += (float)src[(size_t)i * HD_];
    __bf16 r = (__bf16)(s * (1.f / 64.f));
    size_t off = (size_t)bh * 4096 + m * 64 + d;
    if (sel) Klb[off] = r; else Qlb[off] = r;
}

// ---------------- kernel1 = softmax(Q @ K_l^T)  [BH][N][64] bf16 ----------
__global__ __launch_bounds__(256)
void k_kernel1(const __bf16* __restrict__ Qb, const __bf16* __restrict__ Klb,
               __bf16* __restrict__ k1) {
    __shared__ alignas(16) __bf16 Qs[128 * 64];
    __shared__ alignas(16) __bf16 Ks[64 * 64];
    __shared__ float Sf[128 * 64];
    const int tid = threadIdx.x, lane = tid & 31, wave = tid >> 5;
    const int bh = blockIdx.y, chunk = blockIdx.x;
    const __bf16* Qg = Qb + ((size_t)bh * N_ + chunk * 128) * HD_;
    const __bf16* Kg = Klb + (size_t)bh * 4096;
#pragma unroll
    for (int c = tid; c < 1024; c += 256) {
        int row = c >> 3, col = (c & 7) << 3;
        *(uint4*)&Qs[row * 64 + col] = *(const uint4*)&Qg[(size_t)row * 64 + col];
    }
#pragma unroll
    for (int c = tid; c < 512; c += 256) {
        int row = c >> 3, col = (c & 7) << 3;
        *(uint4*)&Ks[row * 64 + col] = *(const uint4*)&Kg[(size_t)row * 64 + col];
    }
    __syncthreads();
    const int m0 = wave * 16;  // each wave: 16 rows x 64 cols = 4 tiles
    v8f acc[4];
#pragma unroll
    for (int j = 0; j < 4; ++j) acc[j] = v8f_zero();
#pragma unroll
    for (int ks = 0; ks < 64; ks += 32) {
        v16bf a = ldsA16x32(&Qs[m0 * 64 + ks], 64);
#pragma unroll
        for (int j = 0; j < 4; ++j) {
            v16bf b = ldsB32x16(&Ks[j * 16 * 64 + ks], 64);
            acc[j] = wmma_bf16(a, b, acc[j]);
        }
    }
    const int er = (lane >> 4) << 3, ec = lane & 15;
#pragma unroll
    for (int j = 0; j < 4; ++j)
#pragma unroll
        for (int e = 0; e < 8; ++e)
            Sf[(m0 + er + e) * 64 + j * 16 + ec] = acc[j][e];
    __syncthreads();
    if (tid < 128) {  // fused row softmax over M=64
        float* row = &Sf[tid * 64];
        float mx = -1e30f;
        for (int m = 0; m < 64; ++m) mx = fmaxf(mx, row[m]);
        float s = 0.f;
        for (int m = 0; m < 64; ++m) { float p = __expf(row[m] - mx); row[m] = p; s += p; }
        float inv = 1.f / s;
        __bf16* dst = k1 + ((size_t)bh * N_ + chunk * 128 + tid) * 64;
        for (int m = 0; m < 64; ++m) dst[m] = (__bf16)(row[m] * inv);
    }
}

// ------- kernel2 softmax + Newton-Schulz inverse (fp32, LDS), store inv2^T bf16
__global__ __launch_bounds__(256)
void k_inv(const __bf16* __restrict__ Qlb, const __bf16* __restrict__ Klb,
           __bf16* __restrict__ inv2T) {
    __shared__ float Af[64 * 64];
    __shared__ float Tf[64 * 64];
    __shared__ float Vf[64 * 64];
    __shared__ __bf16 Ksb[64 * 64];
    __shared__ float red[64];
    __shared__ float denomS;
    const int tid = threadIdx.x;
    const int bh  = blockIdx.x;
    const __bf16* Qg = Qlb + (size_t)bh * 4096;
    const __bf16* Kg = Klb + (size_t)bh * 4096;
    const int i_  = tid >> 2;          // output row
    const int j0_ = (tid & 3) * 16;    // output col base (16 cols per thread)
    float r[16];

    // S2 = Q_l @ K_l^T (fp32)
#pragma unroll
    for (int t = 0; t < 16; ++t) r[t] = 0.f;
    for (int d = 0; d < 64; ++d) {
        float q = (float)Qg[i_ * 64 + d];
#pragma unroll
        for (int t = 0; t < 16; ++t) r[t] += q * (float)Kg[(j0_ + t) * 64 + d];
    }
#pragma unroll
    for (int t = 0; t < 16; ++t) Af[i_ * 64 + j0_ + t] = r[t];
    __syncthreads();
    if (tid < 64) {  // row softmax
        float* row = &Af[tid * 64];
        float mx = -1e30f;
        for (int j = 0; j < 64; ++j) mx = fmaxf(mx, row[j]);
        float s = 0.f;
        for (int j = 0; j < 64; ++j) { float p = __expf(row[j] - mx); row[j] = p; s += p; }
        float inv = 1.f / s;
        for (int j = 0; j < 64; ++j) row[j] *= inv;
    }
    __syncthreads();
    if (tid < 64) {  // column sums
        float s = 0.f;
        for (int i = 0; i < 64; ++i) s += Af[i * 64 + tid];
        red[tid] = s;
    }
    __syncthreads();
    if (tid == 0) {
        float mx = red[0];
        for (int j = 1; j < 64; ++j) mx = fmaxf(mx, red[j]);
        denomS = mx;
    }
    __syncthreads();
    {   // V0 = K^T / denom ; bf16 copy of K for the K@V products
        float invd = 1.f / denomS;
#pragma unroll
        for (int t = 0; t < 16; ++t) {
            int f = tid * 16 + t, i = f >> 6, j = f & 63;
            Vf[f]  = Af[j * 64 + i] * invd;
            Ksb[f] = (__bf16)Af[f];
        }
    }
    __syncthreads();

    for (int it = 0; it < 6; ++it) {
        // A = K @ V
#pragma unroll
        for (int t = 0; t < 16; ++t) r[t] = 0.f;
        for (int d = 0; d < 64; ++d) {
            float kv = (float)Ksb[i_ * 64 + d];
            const float* vr = &Vf[d * 64 + j0_];
#pragma unroll
            for (int t = 0; t < 16; ++t) r[t] += kv * vr[t];
        }
        __syncthreads();
#pragma unroll
        for (int t = 0; t < 16; ++t) Af[i_ * 64 + j0_ + t] = r[t];
        __syncthreads();
        // T = 7I - A
#pragma unroll
        for (int t = 0; t < 16; ++t) {
            int j = j0_ + t;
            Tf[i_ * 64 + j] = (i_ == j ? 7.f : 0.f) - Af[i_ * 64 + j];
        }
        __syncthreads();
        // T = 15I - A @ T
#pragma unroll
        for (int t = 0; t < 16; ++t) r[t] = 0.f;
        for (int d = 0; d < 64; ++d) {
            float a = Af[i_ * 64 + d];
            const float* tr = &Tf[d * 64 + j0_];
#pragma unroll
            for (int t = 0; t < 16; ++t) r[t] += a * tr[t];
        }
        __syncthreads();
#pragma unroll
        for (int t = 0; t < 16; ++t) {
            int j = j0_ + t;
            Tf[i_ * 64 + j] = (i_ == j ? 15.f : 0.f) - r[t];
        }
        __syncthreads();
        // T = 13I - A @ T
#pragma unroll
        for (int t = 0; t < 16; ++t) r[t] = 0.f;
        for (int d = 0; d < 64; ++d) {
            float a = Af[i_ * 64 + d];
            const float* tr = &Tf[d * 64 + j0_];
#pragma unroll
            for (int t = 0; t < 16; ++t) r[t] += a * tr[t];
        }
        __syncthreads();
#pragma unroll
        for (int t = 0; t < 16; ++t) {
            int j = j0_ + t;
            Tf[i_ * 64 + j] = (i_ == j ? 13.f : 0.f) - r[t];
        }
        __syncthreads();
        // V = 0.25 * V @ T
#pragma unroll
        for (int t = 0; t < 16; ++t) r[t] = 0.f;
        for (int d = 0; d < 64; ++d) {
            float v = Vf[i_ * 64 + d];
            const float* tr = &Tf[d * 64 + j0_];
#pragma unroll
            for (int t = 0; t < 16; ++t) r[t] += v * tr[t];
        }
        __syncthreads();
#pragma unroll
        for (int t = 0; t < 16; ++t) Vf[i_ * 64 + j0_ + t] = 0.25f * r[t];
        __syncthreads();
    }
    // store transposed (B-operand layout [N][K]) bf16
#pragma unroll
    for (int t = 0; t < 16; ++t) {
        int j = j0_ + t;
        inv2T[(size_t)bh * 4096 + j * 64 + i_] = (__bf16)Vf[i_ * 64 + j];
    }
}

// -------- fused softmax(Q_l @ K^T) @ V, flash-style over N; output (k3@V)^T ----
__global__ __launch_bounds__(256)
void k_k3v(const __bf16* __restrict__ Qlb, const __bf16* __restrict__ Kb,
           const __bf16* __restrict__ Vb, __bf16* __restrict__ k3vT) {
    __shared__ alignas(16) __bf16 Qls[64 * 64];
    __shared__ alignas(16) __bf16 Kc[64 * 64];
    __shared__ alignas(16) __bf16 Vt[64 * 64];   // V chunk transposed [d][n]
    __shared__ alignas(16) __bf16 Pb[64 * 64];
    __shared__ float Sf[64 * 64];
    __shared__ float rowmax[64], rowsum[64], corr[64];
    const int tid = threadIdx.x, lane = tid & 31, wave = tid >> 5;
    const int bh = blockIdx.x;
    const __bf16* Qg = Qlb + (size_t)bh * 4096;
#pragma unroll
    for (int c = tid; c < 512; c += 256) {
        int row = c >> 3, col = (c & 7) << 3;
        *(uint4*)&Qls[row * 64 + col] = *(const uint4*)&Qg[row * 64 + col];
    }
    if (tid < 64) { rowmax[tid] = -1e30f; rowsum[tid] = 0.f; }
    const int mt  = wave >> 1;        // landmark tile 0..3
    const int dt0 = (wave & 1) * 2;   // chunk-col tile base 0/2
    const int er = (lane >> 4) << 3, ec = lane & 15;
    v8f acc[2]; acc[0] = v8f_zero(); acc[1] = v8f_zero();
    __syncthreads();

    for (int c0 = 0; c0 < N_; c0 += 64) {
        const __bf16* Kg = Kb + ((size_t)bh * N_ + c0) * HD_;
        const __bf16* Vg = Vb + ((size_t)bh * N_ + c0) * HD_;
#pragma unroll
        for (int c = tid; c < 512; c += 256) {
            int row = c >> 3, col = (c & 7) << 3;
            *(uint4*)&Kc[row * 64 + col] = *(const uint4*)&Kg[(size_t)row * 64 + col];
        }
        {   // transpose V chunk into LDS
            int n = tid >> 2, db = (tid & 3) * 16;
#pragma unroll
            for (int t = 0; t < 16; ++t)
                Vt[(db + t) * 64 + n] = Vg[(size_t)n * 64 + db + t];
        }
        __syncthreads();
        // S = Q_l @ Kc^T  (64x64)
        v8f sa[2]; sa[0] = v8f_zero(); sa[1] = v8f_zero();
#pragma unroll
        for (int ks = 0; ks < 64; ks += 32) {
            v16bf a = ldsA16x32(&Qls[mt * 16 * 64 + ks], 64);
#pragma unroll
            for (int t = 0; t < 2; ++t) {
                v16bf b = ldsB32x16(&Kc[(dt0 + t) * 16 * 64 + ks], 64);
                sa[t] = wmma_bf16(a, b, sa[t]);
            }
        }
#pragma unroll
        for (int t = 0; t < 2; ++t)
#pragma unroll
            for (int e = 0; e < 8; ++e)
                Sf[(mt * 16 + er + e) * 64 + (dt0 + t) * 16 + ec] = sa[t][e];
        __syncthreads();
        if (tid < 64) {   // online softmax update for row m=tid
            float* row = &Sf[tid * 64];
            float mx = rowmax[tid], cm = -1e30f;
            for (int n = 0; n < 64; ++n) cm = fmaxf(cm, row[n]);
            float nm = fmaxf(mx, cm);
            float cr = __expf(mx - nm);
            float s = 0.f;
            __bf16* pr = &Pb[tid * 64];
            for (int n = 0; n < 64; ++n) {
                float p = __expf(row[n] - nm);
                pr[n] = (__bf16)p; s += p;
            }
            rowsum[tid] = rowsum[tid] * cr + s;
            rowmax[tid] = nm;
            corr[tid]   = cr;
        }
        __syncthreads();
#pragma unroll
        for (int t = 0; t < 2; ++t)
#pragma unroll
            for (int e = 0; e < 8; ++e) acc[t][e] *= corr[mt * 16 + er + e];
#pragma unroll
        for (int ks = 0; ks < 64; ks += 32) {
            v16bf a = ldsA16x32(&Pb[mt * 16 * 64 + ks], 64);
#pragma unroll
            for (int t = 0; t < 2; ++t) {
                v16bf b = ldsB32x16(&Vt[(dt0 + t) * 16 * 64 + ks], 64);
                acc[t] = wmma_bf16(a, b, acc[t]);
            }
        }
        __syncthreads();
    }
#pragma unroll
    for (int t = 0; t < 2; ++t)
#pragma unroll
        for (int e = 0; e < 8; ++e) {
            int m = mt * 16 + er + e;
            int d = (dt0 + t) * 16 + ec;
            float v = acc[t][e] / rowsum[m];
            k3vT[(size_t)bh * 4096 + d * 64 + m] = (__bf16)v;   // [N=d][K=m]
        }
}

// -------- SV = (k1 @ inv2) @ (k3@V) ; write fp32 SV (head-merged) + bf16 copy --
__global__ __launch_bounds__(256)
void k_sv(const __bf16* __restrict__ k1, const __bf16* __restrict__ inv2T,
          const __bf16* __restrict__ k3vT, float* __restrict__ svOut,
          __bf16* __restrict__ SVb) {
    __shared__ alignas(16) __bf16 K1s[128 * 64];
    __shared__ alignas(16) __bf16 Ti[64 * 64];
    __shared__ alignas(16) __bf16 Gi[64 * 64];
    __shared__ alignas(16) __bf16 Tb[128 * 64];
    const int tid = threadIdx.x, lane = tid & 31, wave = tid >> 5;
    const int bh = blockIdx.y, chunk = blockIdx.x;
    const __bf16* K1g = k1 + ((size_t)bh * N_ + chunk * 128) * 64;
#pragma unroll
    for (int c = tid; c < 1024; c += 256) {
        int row = c >> 3, col = (c & 7) << 3;
        *(uint4*)&K1s[row * 64 + col] = *(const uint4*)&K1g[(size_t)row * 64 + col];
    }
#pragma unroll
    for (int c = tid; c < 512; c += 256) {
        int row = c >> 3, col = (c & 7) << 3;
        *(uint4*)&Ti[row * 64 + col] = *(const uint4*)&inv2T[(size_t)bh * 4096 + row * 64 + col];
        *(uint4*)&Gi[row * 64 + col] = *(const uint4*)&k3vT[(size_t)bh * 4096 + row * 64 + col];
    }
    __syncthreads();
    const int m0 = wave * 16;
    const int er = (lane >> 4) << 3, ec = lane & 15;
    v8f acc[4];
#pragma unroll
    for (int j = 0; j < 4; ++j) acc[j] = v8f_zero();
#pragma unroll
    for (int ks = 0; ks < 64; ks += 32) {
        v16bf a = ldsA16x32(&K1s[m0 * 64 + ks], 64);
#pragma unroll
        for (int j = 0; j < 4; ++j) {
            v16bf b = ldsB32x16(&Ti[j * 16 * 64 + ks], 64);
            acc[j] = wmma_bf16(a, b, acc[j]);
        }
    }
#pragma unroll
    for (int j = 0; j < 4; ++j)
#pragma unroll
        for (int e = 0; e < 8; ++e)
            Tb[(m0 + er + e) * 64 + j * 16 + ec] = (__bf16)acc[j][e];
    __syncthreads();
#pragma unroll
    for (int j = 0; j < 4; ++j) acc[j] = v8f_zero();
#pragma unroll
    for (int ks = 0; ks < 64; ks += 32) {
        v16bf a = ldsA16x32(&Tb[m0 * 64 + ks], 64);
#pragma unroll
        for (int j = 0; j < 4; ++j) {
            v16bf b = ldsB32x16(&Gi[j * 16 * 64 + ks], 64);
            acc[j] = wmma_bf16(a, b, acc[j]);
        }
    }
    const int b = bh >> 4, h = bh & 15;
#pragma unroll
    for (int j = 0; j < 4; ++j)
#pragma unroll
        for (int e = 0; e < 8; ++e) {
            int n = chunk * 128 + m0 + er + e;
            int d = j * 16 + ec;
            size_t off = ((size_t)b * N_ + n) * DIM_ + h * HD_ + d;
            float v = acc[j][e];
            svOut[off] = v;          // second tuple output (fp32)
            SVb[off]   = (__bf16)v;  // feed final projection
        }
}

// ---------------- launcher ----------------
extern "C" void kernel_launch(void* const* d_in, const int* in_sizes, int n_in,
                              void* d_out, int out_size, void* d_ws, size_t ws_size,
                              hipStream_t stream) {
    const float* x      = (const float*)d_in[0];
    const float* w_qkv  = (const float*)d_in[1];
    const float* w_proj = (const float*)d_in[2];
    float* outF = (float*)d_out;                       // [16384][1024] fp32
    float* svF  = outF + (size_t)BN_ * DIM_;           // [16384][1024] fp32

    char* p = (char*)d_ws;
    auto alloc = [&](size_t bytes) -> void* {
        void* r = (void*)p;
        p += (bytes + 255) & ~(size_t)255;
        return r;
    };
    __bf16* Xb     = (__bf16*)alloc((size_t)BN_ * DIM_ * 2);   // x, bf16
    __bf16* WqkvT  = (__bf16*)alloc((size_t)K3_ * DIM_ * 2);   // [3072][1024]
    __bf16* WprojT = (__bf16*)alloc((size_t)DIM_ * DIM_ * 2);  // [1024][1024]
    __bf16* Qb     = (__bf16*)alloc((size_t)BH_ * N_ * HD_ * 2);
    __bf16* Kb     = (__bf16*)alloc((size_t)BH_ * N_ * HD_ * 2);
    __bf16* Vb     = (__bf16*)alloc((size_t)BH_ * N_ * HD_ * 2);
    __bf16* Qlb    = (__bf16*)alloc((size_t)BH_ * 64 * 64 * 2);
    __bf16* Klb    = (__bf16*)alloc((size_t)BH_ * 64 * 64 * 2);
    __bf16* k1     = (__bf16*)alloc((size_t)BH_ * N_ * 64 * 2);
    __bf16* inv2T  = (__bf16*)alloc((size_t)BH_ * 64 * 64 * 2);
    __bf16* k3vT   = (__bf16*)alloc((size_t)BH_ * 64 * 64 * 2);
    __bf16* SVb    = (__bf16*)alloc((size_t)BN_ * DIM_ * 2);

    k_cvt<<<(BN_ * DIM_) / 256, 256, 0, stream>>>(x, Xb, BN_ * DIM_);
    k_cvt_T<<<(K3_ * DIM_) / 256, 256, 0, stream>>>(w_qkv, WqkvT, DIM_, K3_);
    k_cvt_T<<<(DIM_ * DIM_) / 256, 256, 0, stream>>>(w_proj, WprojT, DIM_, DIM_);

    k_gemm128<<<dim3(K3_ / 128, BN_ / 128), 256, 0, stream>>>(
        Xb, WqkvT, DIM_, 0, Qb, Kb, Vb, nullptr);

    k_landmark<<<2048, 256, 0, stream>>>(Qb, Kb, Qlb, Klb);

    k_kernel1<<<dim3(N_ / 128, BH_), 256, 0, stream>>>(Qb, Klb, k1);

    k_inv<<<BH_, 256, 0, stream>>>(Qlb, Klb, inv2T);

    k_k3v<<<BH_, 256, 0, stream>>>(Qlb, Kb, Vb, k3vT);

    k_sv<<<dim3(N_ / 128, BH_), 256, 0, stream>>>(k1, inv2T, k3vT, svF, SVb);

    k_gemm128<<<dim3(DIM_ / 128, BN_ / 128), 256, 0, stream>>>(
        SVb, WprojT, DIM_, 1, nullptr, nullptr, nullptr, outF);
}
